// PointNetSetAbstraction_78228534329994
// MI455X (gfx1250) — matrix operand order repaired and checked by reading, not compiled, not run on hardware
//
#include <hip/hip_runtime.h>

typedef _Float16 h8   __attribute__((ext_vector_type(8)));
typedef _Float16 v16h __attribute__((ext_vector_type(16)));
typedef float    v8f  __attribute__((ext_vector_type(8)));

#define NB    8
#define NPTS  4096
#define NPT   1024
#define NSAMP 32
#define R2    0.04f
#define EPSBN 1e-5f
#define MTOT  262144   // 8*1024*32

// ---------------------------------------------------------------------------
// 1) Farthest point sampling: one workgroup (512 thr = 16 waves) per batch.
//    Points live in registers (8/thread); argmax via wave shuffle + LDS.
// ---------------------------------------------------------------------------
__global__ __launch_bounds__(512) void fps_kernel(const float* __restrict__ xyz,
                                                  int* __restrict__ centIdx) {
  const int b = blockIdx.x;
  const int tid = threadIdx.x;
  const float* X = xyz + (size_t)b * NPTS * 3;

  float px[8], py[8], pz[8], dist[8];
  #pragma unroll
  for (int j = 0; j < 8; ++j) {
    int p = j * 512 + tid;
    px[j] = X[p * 3 + 0]; py[j] = X[p * 3 + 1]; pz[j] = X[p * 3 + 2];
    dist[j] = 1e10f;
  }

  __shared__ float s_v[16];
  __shared__ int   s_i[16];
  __shared__ int   s_best;

  const int lane = tid & 31, wid = tid >> 5;
  int cur = 0;
  for (int it = 0; it < NPT; ++it) {
    if (tid == 0) centIdx[b * NPT + it] = cur;
    float cx = X[cur * 3 + 0], cy = X[cur * 3 + 1], cz = X[cur * 3 + 2];
    float bv = -1.0f; int bi = 0;
    #pragma unroll
    for (int j = 0; j < 8; ++j) {
      float dx = px[j] - cx, dy = py[j] - cy, dz = pz[j] - cz;
      float d = dx * dx + dy * dy + dz * dz;
      float nd = fminf(dist[j], d);
      dist[j] = nd;
      int p = j * 512 + tid;
      if (nd > bv || (nd == bv && p < bi)) { bv = nd; bi = p; }
    }
    #pragma unroll
    for (int off = 16; off > 0; off >>= 1) {          // wave32 argmax
      float ov = __shfl_xor(bv, off, 32);
      int   oi = __shfl_xor(bi, off, 32);
      if (ov > bv || (ov == bv && oi < bi)) { bv = ov; bi = oi; }
    }
    if (lane == 0) { s_v[wid] = bv; s_i[wid] = bi; }
    __syncthreads();
    if (wid == 0) {
      float v2 = (lane < 16) ? s_v[lane] : -2.0f;
      int   i2 = (lane < 16) ? s_i[lane] : 0;
      #pragma unroll
      for (int off = 16; off > 0; off >>= 1) {
        float ov = __shfl_xor(v2, off, 32);
        int   oi = __shfl_xor(i2, off, 32);
        if (ov > v2 || (ov == v2 && oi < i2)) { v2 = ov; i2 = oi; }
      }
      if (lane == 0) s_best = i2;
    }
    __syncthreads();
    cur = s_best;
  }
}

// ---------------------------------------------------------------------------
// 2) Ball query + grouping: one wave per centroid. Ordered first-32-in-radius
//    selection with ballot/popc; emit f16 WMMA-ready rows (K padded to 96)
//    and new_xyz directly into d_out.
// ---------------------------------------------------------------------------
__global__ __launch_bounds__(256) void group_kernel(
    const float* __restrict__ xyz, const float* __restrict__ pts,
    const int* __restrict__ centIdx, _Float16* __restrict__ X0,
    float* __restrict__ outNewXyz) {
  const int gwave = blockIdx.x * 8 + (threadIdx.x >> 5);   // 0..8191
  const int lane = threadIdx.x & 31;
  const int b = gwave >> 10;
  const int s = gwave & 1023;
  const float* Xb = xyz + (size_t)b * NPTS * 3;

  const int cIdx = centIdx[b * NPT + s];
  const float cx = Xb[cIdx * 3 + 0], cy = Xb[cIdx * 3 + 1], cz = Xb[cIdx * 3 + 2];

  __shared__ int s_sel[8 * 32];
  int* sel = s_sel + (threadIdx.x >> 5) * 32;

  int cnt = 0;                                   // uniform across the wave
  for (int ch = 0; ch < NPTS / 32 && cnt < 32; ++ch) {
    int p = ch * 32 + lane;
    float dx = Xb[p * 3 + 0] - cx, dy = Xb[p * 3 + 1] - cy, dz = Xb[p * 3 + 2] - cz;
    float d = dx * dx + dy * dy + dz * dz;
    bool in = d <= R2;
    unsigned mask = (unsigned)__ballot(in);      // wave32: low 32 bits
    int rank = cnt + __popc(mask & ((1u << lane) - 1u));
    if (in && rank < 32) sel[rank] = p;
    cnt += __popc(mask);
  }
  __syncthreads();
  if (cnt < 32) {                                // pad with first index
    int first = sel[0];
    if (lane >= cnt) sel[lane] = first;
  }
  __syncthreads();

  const int q = sel[lane];
  const size_t m = ((size_t)(b * NPT + s)) * NSAMP + lane;
  _Float16* row = X0 + m * 96;
  row[0] = (_Float16)(Xb[q * 3 + 0] - cx);
  row[1] = (_Float16)(Xb[q * 3 + 1] - cy);
  row[2] = (_Float16)(Xb[q * 3 + 2] - cz);
  const float* P = pts + ((size_t)b * NPTS + q) * 64;
  #pragma unroll 8
  for (int c = 0; c < 64; ++c) row[3 + c] = (_Float16)P[c];
  #pragma unroll
  for (int c = 67; c < 96; ++c) row[c] = (_Float16)0.0f;

  if (lane == 0) {
    float* o = outNewXyz + (size_t)(b * NPT + s) * 3;
    o[0] = cx; o[1] = cy; o[2] = cz;
  }
}

// ---------------------------------------------------------------------------
// 3) Weight conversion fp32 -> f16, K-padded (zero fill).
// ---------------------------------------------------------------------------
__global__ void convw_kernel(const float* __restrict__ W, _Float16* __restrict__ Wp,
                             int O, int C, int Kpad) {
  int i = blockIdx.x * blockDim.x + threadIdx.x;
  if (i >= O * Kpad) return;
  int o = i / Kpad, c = i % Kpad;
  Wp[i] = (c < C) ? (_Float16)W[o * C + c] : (_Float16)0.0f;
}

__global__ void zstats_kernel(float* gSum, float* gSq) {
  int i = threadIdx.x;
  if (i < 128) { gSum[i] = 0.0f; gSq[i] = 0.0f; }
}

// ---------------------------------------------------------------------------
// 4) WMMA GEMM: Y[M,O] = X[M,KPAD](f16) * Wp[O,KPAD]^T(f16) + bias, fp32 acc.
//    One wave per 16-row M-tile computing ALL O/16 N-tiles: each A fragment
//    is read from HBM exactly once (bandwidth-bound regime). Weights are
//    staged once per block into LDS; B fragments come from ds_load.
//    Fused per-channel sum/sumsq epilogue via LDS + global atomics.
//    A-frag: lane(row=lane&15, khi=lane>>4), elems = K khi*16..+15 (contiguous).
//    B-frag: lane(n=lane&15, khi), elems = K khi*16..+15 (contiguous).
//    C/D:    m = e + 8*khi, n = lane&15.
// ---------------------------------------------------------------------------
template <int KPAD, int O>
__global__ __launch_bounds__(256) void gemm_kernel(
    const _Float16* __restrict__ Xin, const _Float16* __restrict__ Wp,
    const float* __restrict__ bias, float* __restrict__ Y,
    float* __restrict__ gSum, float* __restrict__ gSq) {
  constexpr int NT = O / 16;
  __shared__ __align__(16) _Float16 s_w[O * KPAD];
  __shared__ float s_sum[128], s_sq[128];

  const int tid = threadIdx.x;
  if (tid < 128) { s_sum[tid] = 0.0f; s_sq[tid] = 0.0f; }
  // stage the (tiny) weight matrix into LDS, 16B per thread per iter
  {
    const uint4* src = (const uint4*)Wp;
    uint4* dst = (uint4*)s_w;
    #pragma unroll
    for (int i = tid; i < O * KPAD / 8; i += 256) dst[i] = src[i];
  }
  __syncthreads();

  const int wid = tid >> 5, lane = tid & 31;
  const int mBase = (blockIdx.x * 8 + wid) * 16;
  const int khi = lane >> 4;
  const int n = lane & 15;
  const int row = mBase + n;

  v8f acc[NT];
  #pragma unroll
  for (int nt = 0; nt < NT; ++nt) acc[nt] = v8f{};

  #pragma unroll
  for (int kb = 0; kb < KPAD; kb += 32) {
    union { v16h v; h8 h[2]; } af;
    const _Float16* ap = Xin + (size_t)row * KPAD + kb + khi * 16;
    af.h[0] = *(const h8*)(ap);
    af.h[1] = *(const h8*)(ap + 8);
    #pragma unroll
    for (int nt = 0; nt < NT; ++nt) {
      v16h bf = *(const v16h*)(s_w + (nt * 16 + n) * KPAD + kb + khi * 16);
      acc[nt] = __builtin_amdgcn_wmma_f32_16x16x32_f16(false, af.v, false, bf,
                                                       (short)0, acc[nt], false, false);
    }
  }

  #pragma unroll
  for (int nt = 0; nt < NT; ++nt) {
    const int ch = nt * 16 + n;
    const float bv = bias[ch];
    float ps = 0.0f, pq = 0.0f;
    #pragma unroll
    for (int e = 0; e < 8; ++e) {
      float v = acc[nt][e] + bv;
      int m = mBase + e + 8 * khi;
      Y[(size_t)m * O + ch] = v;
      ps += v; pq += v * v;
    }
    atomicAdd(&s_sum[ch], ps);   // ds_add_f32
    atomicAdd(&s_sq[ch], pq);
  }
  __syncthreads();
  if (tid < O) {
    atomicAdd(&gSum[tid], s_sum[tid]);  // global_atomic_add_f32
    atomicAdd(&gSq[tid], s_sq[tid]);
  }
}

// ---------------------------------------------------------------------------
// 5) BN stat finalize + fused BN/ReLU passes.
// ---------------------------------------------------------------------------
__global__ void bnstats_kernel(const float* gSum, const float* gSq,
                               const float* gamma, const float* beta,
                               float* scale, float* shift, int O, float invM) {
  int i = threadIdx.x;
  if (i < O) {
    float mean = gSum[i] * invM;
    float var = gSq[i] * invM - mean * mean;
    float sc = gamma[i] * rsqrtf(var + EPSBN);
    scale[i] = sc;
    shift[i] = beta[i] - mean * sc;
  }
}

__global__ void bnrelu_kernel(const float* __restrict__ Y, const float* __restrict__ scale,
                              const float* __restrict__ shift, _Float16* __restrict__ Xn,
                              int O, size_t total) {
  size_t i = (size_t)blockIdx.x * blockDim.x + threadIdx.x;
  if (i >= total) return;
  int o = (int)(i % O);
  float v = Y[i] * scale[o] + shift[o];
  v = v > 0.0f ? v : 0.0f;
  Xn[i] = (_Float16)v;        // next layer's Kpad == O (64), identity layout
}

__global__ void bnrelumax_kernel(const float* __restrict__ Y, const float* __restrict__ scale,
                                 const float* __restrict__ shift, float* __restrict__ out) {
  int i = blockIdx.x * blockDim.x + threadIdx.x;  // 8192*128
  if (i >= NB * NPT * 128) return;
  int o = i & 127;
  int bs = i >> 7;
  float sc = scale[o], sh = shift[o];
  float mx = -1e30f;
  const float* yp = Y + (size_t)bs * NSAMP * 128 + o;
  #pragma unroll 4
  for (int k = 0; k < NSAMP; ++k) {
    float v = yp[k * 128] * sc + sh;
    v = v > 0.0f ? v : 0.0f;
    mx = v > mx ? v : mx;
  }
  out[i] = mx;
}

// ---------------------------------------------------------------------------
extern "C" void kernel_launch(void* const* d_in, const int* in_sizes, int n_in,
                              void* d_out, int out_size, void* d_ws, size_t ws_size,
                              hipStream_t stream) {
  const float* xyz = (const float*)d_in[0];
  const float* pts = (const float*)d_in[1];
  const float* W0 = (const float*)d_in[2];  const float* b0 = (const float*)d_in[3];
  const float* g0 = (const float*)d_in[4];  const float* be0 = (const float*)d_in[5];
  const float* W1 = (const float*)d_in[6];  const float* b1 = (const float*)d_in[7];
  const float* g1 = (const float*)d_in[8];  const float* be1 = (const float*)d_in[9];
  const float* W2 = (const float*)d_in[10]; const float* b2 = (const float*)d_in[11];
  const float* g2 = (const float*)d_in[12]; const float* be2 = (const float*)d_in[13];
  float* out = (float*)d_out;

  char* ws = (char*)d_ws;
  size_t off = 0;
  auto alloc = [&](size_t bytes) -> void* {
    void* p = ws + off;
    off += (bytes + 255) & ~(size_t)255;
    return p;
  };
  int*      centIdx = (int*)alloc((size_t)NB * NPT * 4);
  float*    gSum  = (float*)alloc(512);
  float*    gSq   = (float*)alloc(512);
  float*    scale = (float*)alloc(512);
  float*    shift = (float*)alloc(512);
  _Float16* Wp0 = (_Float16*)alloc((size_t)64 * 96 * 2);
  _Float16* Wp1 = (_Float16*)alloc((size_t)64 * 64 * 2);
  _Float16* Wp2 = (_Float16*)alloc((size_t)128 * 64 * 2);
  _Float16* X0  = (_Float16*)alloc((size_t)MTOT * 96 * 2);   // 48 MB (reused as X2)
  _Float16* X1  = (_Float16*)alloc((size_t)MTOT * 64 * 2);   // 32 MB
  float*    Y   = (float*)alloc((size_t)MTOT * 128 * 4);     // 128 MB
  _Float16* X2  = X0;                                        // alias: X0 dead after L1

  const float invM = 1.0f / (float)MTOT;
  float* outPts = out + (size_t)NB * NPT * 3;
  const unsigned gemmBlocks = MTOT / 16 / 8;                 // 2048

  // stage 1: FPS + grouping + weight prep
  fps_kernel<<<NB, 512, 0, stream>>>(xyz, centIdx);
  convw_kernel<<<(64 * 96 + 255) / 256, 256, 0, stream>>>(W0, Wp0, 64, 67, 96);
  convw_kernel<<<(64 * 64 + 255) / 256, 256, 0, stream>>>(W1, Wp1, 64, 64, 64);
  convw_kernel<<<(128 * 64 + 255) / 256, 256, 0, stream>>>(W2, Wp2, 128, 64, 64);
  group_kernel<<<NB * NPT / 8, 256, 0, stream>>>(xyz, pts, centIdx, X0, out);

  // layer 0: 67(pad96) -> 64
  zstats_kernel<<<1, 128, 0, stream>>>(gSum, gSq);
  gemm_kernel<96, 64><<<gemmBlocks, 256, 0, stream>>>(X0, Wp0, b0, Y, gSum, gSq);
  bnstats_kernel<<<1, 128, 0, stream>>>(gSum, gSq, g0, be0, scale, shift, 64, invM);
  bnrelu_kernel<<<(unsigned)(((size_t)MTOT * 64 + 255) / 256), 256, 0, stream>>>(
      Y, scale, shift, X1, 64, (size_t)MTOT * 64);

  // layer 1: 64 -> 64
  zstats_kernel<<<1, 128, 0, stream>>>(gSum, gSq);
  gemm_kernel<64, 64><<<gemmBlocks, 256, 0, stream>>>(X1, Wp1, b1, Y, gSum, gSq);
  bnstats_kernel<<<1, 128, 0, stream>>>(gSum, gSq, g1, be1, scale, shift, 64, invM);
  bnrelu_kernel<<<(unsigned)(((size_t)MTOT * 64 + 255) / 256), 256, 0, stream>>>(
      Y, scale, shift, X2, 64, (size_t)MTOT * 64);

  // layer 2: 64 -> 128, then max over the 32 samples per group
  zstats_kernel<<<1, 128, 0, stream>>>(gSum, gSq);
  gemm_kernel<64, 128><<<gemmBlocks, 256, 0, stream>>>(X2, Wp2, b2, Y, gSum, gSq);
  bnstats_kernel<<<1, 128, 0, stream>>>(gSum, gSq, g2, be2, scale, shift, 128, invM);
  bnrelumax_kernel<<<(NB * NPT * 128 + 255) / 256, 256, 0, stream>>>(Y, scale, shift, outPts);
}
